// ClipLoss_25151328485434
// MI455X (gfx1250) — compile-verified, matrix operand order checked by reading, not run on hardware
//
#include <hip/hip_runtime.h>

typedef __attribute__((ext_vector_type(2)))  float  v2f;
typedef __attribute__((ext_vector_type(8)))  float  v8f;
typedef __attribute__((ext_vector_type(4)))  __bf16 v4bf;
typedef __attribute__((ext_vector_type(8)))  __bf16 v8bf;
typedef __attribute__((ext_vector_type(16))) __bf16 v16bf;

#define NROWS 16384
#define DIM   512
#define LOGIT_SCALE (1.0f / 0.07f)

// ---------------------------------------------------------------------------
// Kernel 0: zero the two f64 accumulators (ws is poisoned; re-zero each call).
// ---------------------------------------------------------------------------
__global__ void clip_init_acc(double* __restrict__ acc) {
    if (threadIdx.x == 0) { acc[0] = 0.0; acc[1] = 0.0; }
}

// ---------------------------------------------------------------------------
// Kernel 1: diag[i] = LOGIT_SCALE * <img_i, txt_i>.  One wave per row (fp32).
// ---------------------------------------------------------------------------
__global__ __launch_bounds__(256) void clip_diag(const float* __restrict__ img,
                                                 const float* __restrict__ txt,
                                                 float* __restrict__ diag) {
    const int wave = threadIdx.x >> 5;
    const int lane = threadIdx.x & 31;
    const int row  = blockIdx.x * 8 + wave;
    const float4* a = (const float4*)(img + (size_t)row * DIM);
    const float4* b = (const float4*)(txt + (size_t)row * DIM);
    float s = 0.f;
#pragma unroll
    for (int it = 0; it < 4; ++it) {            // 4 * 32 lanes * 4 floats = 512
        float4 av = a[lane + it * 32];
        float4 bv = b[lane + it * 32];
        s += av.x * bv.x + av.y * bv.y + av.z * bv.z + av.w * bv.w;
    }
#pragma unroll
    for (int m = 16; m >= 1; m >>= 1) s += __shfl_xor(s, m, 32);
    if (lane == 0) diag[row] = LOGIT_SCALE * s;
}

// ---------------------------------------------------------------------------
// Kernel 2: split fp32 -> (hi, lo) bf16 planes: hi = bf16(x), lo = bf16(x-hi).
// ---------------------------------------------------------------------------
__global__ __launch_bounds__(256) void clip_split_bf16(const float* __restrict__ x,
                                                       __bf16* __restrict__ hi,
                                                       __bf16* __restrict__ lo,
                                                       int n) {
    int i = (blockIdx.x * 256 + threadIdx.x) * 4;
    if (i >= n) return;
    float4 v = *(const float4*)(x + i);
    v4bf h, l;
    h.x = (__bf16)v.x; l.x = (__bf16)(v.x - (float)h.x);
    h.y = (__bf16)v.y; l.y = (__bf16)(v.y - (float)h.y);
    h.z = (__bf16)v.z; l.z = (__bf16)(v.z - (float)h.z);
    h.w = (__bf16)v.w; l.w = (__bf16)(v.w - (float)h.w);
    *(v4bf*)(hi + i) = h;
    *(v4bf*)(lo + i) = l;
}

// Load one 16-bit A/B fragment (NT layout): lane l, row = base row + (l&15),
// K chunks [k + 8*(l>>4), +8) and [k + 8*(l>>4) + 16, +8). Pointer is
// pre-offset by row and 8*(l>>4); chunks are p[0..7] and p[16..23].
__device__ __forceinline__ v16bf ldfrag(const __bf16* p) {
    v8bf a = *(const v8bf*)(p);
    v8bf b = *(const v8bf*)(p + 16);
    return __builtin_shufflevector(a, b, 0, 1, 2, 3, 4, 5, 6, 7,
                                   8, 9, 10, 11, 12, 13, 14, 15);
}

// ---------------------------------------------------------------------------
// Shared online-LSE update + epilogue, used by both GEMM variants.
//   acc layout (v_wmma C/D): vgpr v, lane l -> M = v + 8*(l>>4), N = l&15.
// ---------------------------------------------------------------------------
__device__ __forceinline__ void lse_update(const v8f& a0, const v8f& a1,
                                           const v8f& a2, const v8f& a3,
                                           float rm[8], float rs[8]) {
#pragma unroll
    for (int v = 0; v < 8; ++v) {
        float x0 = LOGIT_SCALE * a0[v];
        float x1 = LOGIT_SCALE * a1[v];
        float x2 = LOGIT_SCALE * a2[v];
        float x3 = LOGIT_SCALE * a3[v];
        float lm = fmaxf(fmaxf(x0, x1), fmaxf(x2, x3));
        lm = fmaxf(lm, __shfl_xor(lm, 1, 32));
        lm = fmaxf(lm, __shfl_xor(lm, 2, 32));
        lm = fmaxf(lm, __shfl_xor(lm, 4, 32));
        lm = fmaxf(lm, __shfl_xor(lm, 8, 32));
        float nm = fmaxf(rm[v], lm);
        float p  = __expf(x0 - nm) + __expf(x1 - nm) +
                   __expf(x2 - nm) + __expf(x3 - nm);
        p += __shfl_xor(p, 1, 32);
        p += __shfl_xor(p, 2, 32);
        p += __shfl_xor(p, 4, 32);
        p += __shfl_xor(p, 8, 32);
        rs[v] = rs[v] * __expf(rm[v] - nm) + p;
        rm[v] = nm;
    }
}

__device__ __forceinline__ void lse_epilogue(const float rm[8], const float rs[8],
                                             int r0, int half, int l16,
                                             const float* __restrict__ diag,
                                             double* __restrict__ acc) {
    if (l16 == 0) {
        float local = 0.f;
#pragma unroll
        for (int v = 0; v < 8; ++v) {
            int row = r0 + v + 8 * half;
            local += rm[v] + logf(rs[v]) - diag[row];
        }
        atomicAdd(acc, (double)local);
    }
}

// ---------------------------------------------------------------------------
// Kernel 3a: fused split-bf16 GEMM + online row-LSE for L = scale * (A @ B^T).
//   A ~ Ahi+Alo, B ~ Bhi+Blo;  A*B ~= Ahi*Bhi + Ahi*Blo + Alo*Bhi.
//   Wave owns 16 rows x 64-col strips; K loop of v_wmma_f32_16x16x32_bf16.
// ---------------------------------------------------------------------------
__global__ __launch_bounds__(256) void
clip_lse_pass_bf16(const __bf16* __restrict__ Ahi, const __bf16* __restrict__ Alo,
                   const __bf16* __restrict__ Bhi, const __bf16* __restrict__ Blo,
                   const float* __restrict__ diag, double* __restrict__ acc) {
    const int lane = threadIdx.x & 31;
    const int wave = threadIdx.x >> 5;
    const int half = lane >> 4;
    const int l16  = lane & 15;
    const int r0   = blockIdx.x * 128 + wave * 16;

    float rm[8], rs[8];
#pragma unroll
    for (int v = 0; v < 8; ++v) { rm[v] = -__builtin_inff(); rs[v] = 0.f; }

    const size_t roff = (size_t)(r0 + l16) * DIM + 8 * half;
    const __bf16* ah = Ahi + roff;
    const __bf16* al = Alo + roff;

    for (int c0 = 0; c0 < NROWS; c0 += 64) {
        v8f acc0 = {}, acc1 = {}, acc2 = {}, acc3 = {};
        const size_t c = (size_t)(c0 + l16) * DIM + 8 * half;
        const __bf16* bh0 = Bhi + c;            const __bf16* bl0 = Blo + c;
        const __bf16* bh1 = bh0 + 16 * DIM;     const __bf16* bl1 = bl0 + 16 * DIM;
        const __bf16* bh2 = bh0 + 32 * DIM;     const __bf16* bl2 = bl0 + 32 * DIM;
        const __bf16* bh3 = bh0 + 48 * DIM;     const __bf16* bl3 = bl0 + 48 * DIM;
#pragma unroll 2
        for (int k = 0; k < DIM; k += 32) {
            v16bf fah = ldfrag(ah + k);
            v16bf fal = ldfrag(al + k);
            v16bf b0h = ldfrag(bh0 + k), b0l = ldfrag(bl0 + k);
            v16bf b1h = ldfrag(bh1 + k), b1l = ldfrag(bl1 + k);
            v16bf b2h = ldfrag(bh2 + k), b2l = ldfrag(bl2 + k);
            v16bf b3h = ldfrag(bh3 + k), b3l = ldfrag(bl3 + k);
            acc0 = __builtin_amdgcn_wmma_f32_16x16x32_bf16(false, fal, false, b0h, (short)0, acc0, false, false);
            acc0 = __builtin_amdgcn_wmma_f32_16x16x32_bf16(false, fah, false, b0l, (short)0, acc0, false, false);
            acc0 = __builtin_amdgcn_wmma_f32_16x16x32_bf16(false, fah, false, b0h, (short)0, acc0, false, false);
            acc1 = __builtin_amdgcn_wmma_f32_16x16x32_bf16(false, fal, false, b1h, (short)0, acc1, false, false);
            acc1 = __builtin_amdgcn_wmma_f32_16x16x32_bf16(false, fah, false, b1l, (short)0, acc1, false, false);
            acc1 = __builtin_amdgcn_wmma_f32_16x16x32_bf16(false, fah, false, b1h, (short)0, acc1, false, false);
            acc2 = __builtin_amdgcn_wmma_f32_16x16x32_bf16(false, fal, false, b2h, (short)0, acc2, false, false);
            acc2 = __builtin_amdgcn_wmma_f32_16x16x32_bf16(false, fah, false, b2l, (short)0, acc2, false, false);
            acc2 = __builtin_amdgcn_wmma_f32_16x16x32_bf16(false, fah, false, b2h, (short)0, acc2, false, false);
            acc3 = __builtin_amdgcn_wmma_f32_16x16x32_bf16(false, fal, false, b3h, (short)0, acc3, false, false);
            acc3 = __builtin_amdgcn_wmma_f32_16x16x32_bf16(false, fah, false, b3l, (short)0, acc3, false, false);
            acc3 = __builtin_amdgcn_wmma_f32_16x16x32_bf16(false, fah, false, b3h, (short)0, acc3, false, false);
        }
        lse_update(acc0, acc1, acc2, acc3, rm, rs);
    }
    lse_epilogue(rm, rs, r0, half, l16, diag, acc);
}

// ---------------------------------------------------------------------------
// Kernel 3b: exact-fp32 fallback (v_wmma_f32_16x16x4_f32), used when ws is
// too small for the bf16 split planes.
// ---------------------------------------------------------------------------
__global__ __launch_bounds__(256) void
clip_lse_pass_f32(const float* __restrict__ A, const float* __restrict__ B,
                  const float* __restrict__ diag, double* __restrict__ acc) {
    const int lane = threadIdx.x & 31;
    const int wave = threadIdx.x >> 5;
    const int half = lane >> 4;
    const int l16  = lane & 15;
    const int r0   = blockIdx.x * 128 + wave * 16;

    float rm[8], rs[8];
#pragma unroll
    for (int v = 0; v < 8; ++v) { rm[v] = -__builtin_inff(); rs[v] = 0.f; }

    const float* arow = A + (size_t)(r0 + l16) * DIM + 2 * half;

    for (int c0 = 0; c0 < NROWS; c0 += 64) {
        v8f acc0 = {}, acc1 = {}, acc2 = {}, acc3 = {};
        const float* b0 = B + (size_t)(c0 +  0 + l16) * DIM + 2 * half;
        const float* b1 = B + (size_t)(c0 + 16 + l16) * DIM + 2 * half;
        const float* b2 = B + (size_t)(c0 + 32 + l16) * DIM + 2 * half;
        const float* b3 = B + (size_t)(c0 + 48 + l16) * DIM + 2 * half;
#pragma unroll 4
        for (int k = 0; k < DIM; k += 4) {
            v2f af = *(const v2f*)(arow + k);
            v2f f0 = *(const v2f*)(b0 + k);
            v2f f1 = *(const v2f*)(b1 + k);
            v2f f2 = *(const v2f*)(b2 + k);
            v2f f3 = *(const v2f*)(b3 + k);
            acc0 = __builtin_amdgcn_wmma_f32_16x16x4_f32(false, af, false, f0, (short)0, acc0, false, false);
            acc1 = __builtin_amdgcn_wmma_f32_16x16x4_f32(false, af, false, f1, (short)0, acc1, false, false);
            acc2 = __builtin_amdgcn_wmma_f32_16x16x4_f32(false, af, false, f2, (short)0, acc2, false, false);
            acc3 = __builtin_amdgcn_wmma_f32_16x16x4_f32(false, af, false, f3, (short)0, acc3, false, false);
        }
        lse_update(acc0, acc1, acc2, acc3, rm, rs);
    }
    lse_epilogue(rm, rs, r0, half, l16, diag, acc);
}

// ---------------------------------------------------------------------------
// Kernel 4: out = (acc0 + acc1) / (2N)
// ---------------------------------------------------------------------------
__global__ void clip_finalize(const double* __restrict__ acc, float* __restrict__ out) {
    if (threadIdx.x == 0)
        out[0] = (float)((acc[0] + acc[1]) / (2.0 * (double)NROWS));
}

// ---------------------------------------------------------------------------
extern "C" void kernel_launch(void* const* d_in, const int* in_sizes, int n_in,
                              void* d_out, int out_size, void* d_ws, size_t ws_size,
                              hipStream_t stream) {
    const float* img = (const float*)d_in[0];
    const float* txt = (const float*)d_in[1];
    float* out = (float*)d_out;

    const size_t plane = (size_t)NROWS * DIM;                 // elements
    float*  diag    = (float*)d_ws;
    size_t  off_acc = ((size_t)NROWS * sizeof(float) + 63) & ~(size_t)63;
    double* acc     = (double*)((char*)d_ws + off_acc);
    size_t  off_bf  = off_acc + 64;
    __bf16* Ahi = (__bf16*)((char*)d_ws + off_bf);
    __bf16* Alo = Ahi + plane;
    __bf16* Bhi = Alo + plane;
    __bf16* Blo = Bhi + plane;
    const size_t need = off_bf + 4 * plane * sizeof(__bf16);

    clip_init_acc<<<1, 64, 0, stream>>>(acc);
    clip_diag<<<NROWS / 8, 256, 0, stream>>>(img, txt, diag);

    if (ws_size >= need) {
        const int n = (int)plane;                              // 8,388,608
        const int blocks = (n / 4 + 255) / 256;
        clip_split_bf16<<<blocks, 256, 0, stream>>>(img, Ahi, Alo, n);
        clip_split_bf16<<<blocks, 256, 0, stream>>>(txt, Bhi, Blo, n);
        clip_lse_pass_bf16<<<NROWS / 128, 256, 0, stream>>>(Ahi, Alo, Bhi, Blo, diag, &acc[0]);
        clip_lse_pass_bf16<<<NROWS / 128, 256, 0, stream>>>(Bhi, Blo, Ahi, Alo, diag, &acc[1]);
    } else {
        clip_lse_pass_f32<<<NROWS / 128, 256, 0, stream>>>(img, txt, diag, &acc[0]);
        clip_lse_pass_f32<<<NROWS / 128, 256, 0, stream>>>(txt, img, diag, &acc[1]);
    }
    clip_finalize<<<1, 64, 0, stream>>>(acc, out);
}
